// PyGAbAgIntGAE_6983616823351
// MI455X (gfx1250) — matrix operand order repaired and checked by reading, not compiled, not run on hardware
//
#include <hip/hip_runtime.h>
#include <cmath>

// ---------------------------------------------------------------------------
// CDNA5 (gfx1250) WMMA types
// ---------------------------------------------------------------------------
typedef __attribute__((ext_vector_type(16))) _Float16 v16h;
typedef __attribute__((ext_vector_type(8)))  float    v8f;
typedef __attribute__((ext_vector_type(2)))  _Float16 h2;

__device__ __forceinline__ float sigm_(float x)     { return 1.0f / (1.0f + expf(-x)); }
__device__ __forceinline__ float gelu_erf_(float x) { return 0.5f * x * (1.0f + erff(x * 0.70710678118654752f)); }

// ---------------------------------------------------------------------------
// Generic batched GEMM:  C = epilogue(alpha * A @ B (+bias))
//   A: [M,K] row-major (lda), B: [K,N] (ldb) or transB -> B:[N,K] row-major
//   batch z in grid.z; z = bb*nh + hh; per-operand offsets bb*s?b + hh*s?h
//   ep: 0 none, 1 +bias, 2 tanh(+bias), 3 gelu(+bias), 4 sigmoid
// Tiles 64x64, 128 threads (4 waves), each wave 2x2 fragments of 16x16,
// K-step 32 through LDS f16 staging, v_wmma_f32_16x16x32_f16 inner loop.
// FULL-TILE CONTRACT: M,N multiples of 64; K multiple of 32; lda/ldb/ldc and
// all batch offsets multiples of 4 floats; base pointers 16B aligned.
// ---------------------------------------------------------------------------
__global__ __launch_bounds__(128)
void gemm_f16wmma(const float* __restrict__ Ab, const float* __restrict__ Bb,
                  const float* __restrict__ bias, float* __restrict__ Cb,
                  int M, int N, int K, int lda, int ldb, int ldc,
                  long sAb, long sAh, long sBb, long sBh, long sCb, long sCh,
                  int nh, int transB, float alpha, int ep)
{
  __shared__ _Float16 As[64][40];   // [m][k] pad 8 halves
  __shared__ _Float16 Bs[64][40];   // [n][k] pad 8 halves

  const int tid  = threadIdx.x;
  const int lane = tid & 31;
  const int wave = tid >> 5;

  const int z  = blockIdx.z;
  const int bb = z / nh, hh = z % nh;
  const float* A = Ab + (long)bb * sAb + (long)hh * sAh;
  const float* B = Bb + (long)bb * sBb + (long)hh * sBh;
  float*       C = Cb + (long)bb * sCb + (long)hh * sCh;

  const int mBase = blockIdx.y * 64;
  const int nBase = blockIdx.x * 64;

  const int mW  = (wave >> 1) * 32;
  const int nW  = (wave & 1) * 32;
  const int fm  = lane & 15;
  const int khi = lane >> 4;

  v8f acc[2][2] = {};

  // staging coordinates (full tiles, vectorized b128 loads)
  const int arow = tid >> 1;
  const int aks  = (tid & 1) * 16;
  const int bkk  = tid >> 2;            // 0..31  (!transB path)
  const int bns  = (tid & 3) * 16;

  for (int kc = 0; kc < K; kc += 32) {
    // ---- stage A tile (f32 -> f16), 4 x b128 loads
    {
      const float4* ap = (const float4*)(A + (long)(mBase + arow) * lda + kc + aks);
#pragma unroll
      for (int i = 0; i < 4; ++i) {
        float4 f = ap[i];
        h2 p0; p0.x = (_Float16)f.x; p0.y = (_Float16)f.y;
        h2 p1; p1.x = (_Float16)f.z; p1.y = (_Float16)f.w;
        *(h2*)&As[arow][aks + 4 * i]     = p0;
        *(h2*)&As[arow][aks + 4 * i + 2] = p1;
      }
    }
    // ---- stage B tile into Bs[n][k]
    if (transB) {
      const float4* bp = (const float4*)(B + (long)(nBase + arow) * ldb + kc + aks);
#pragma unroll
      for (int i = 0; i < 4; ++i) {
        float4 f = bp[i];
        h2 p0; p0.x = (_Float16)f.x; p0.y = (_Float16)f.y;
        h2 p1; p1.x = (_Float16)f.z; p1.y = (_Float16)f.w;
        *(h2*)&Bs[arow][aks + 4 * i]     = p0;
        *(h2*)&Bs[arow][aks + 4 * i + 2] = p1;
      }
    } else {
      const float4* bp = (const float4*)(B + (long)(kc + bkk) * ldb + nBase + bns);
#pragma unroll
      for (int i = 0; i < 4; ++i) {
        float4 f = bp[i];
        Bs[bns + 4 * i + 0][bkk] = (_Float16)f.x;
        Bs[bns + 4 * i + 1][bkk] = (_Float16)f.y;
        Bs[bns + 4 * i + 2][bkk] = (_Float16)f.z;
        Bs[bns + 4 * i + 3][bkk] = (_Float16)f.w;
      }
    }
    __syncthreads();

    // ---- gather fragments per ISA 7.12.2 layouts
    v16h af[2], bf[2];
#pragma unroll
    for (int i = 0; i < 2; ++i) {
#pragma unroll
      for (int v = 0; v < 8; ++v) {
        int k0 = ((v >> 2) << 4) + (khi << 3) + ((v & 3) << 1);
        af[i][2 * v]     = As[mW + 16 * i + fm][k0];
        af[i][2 * v + 1] = As[mW + 16 * i + fm][k0 + 1];
      }
    }
#pragma unroll
    for (int j = 0; j < 2; ++j) {
#pragma unroll
      for (int v = 0; v < 8; ++v) {
        int k0 = ((v >> 2) << 4) + (khi << 3) + ((v & 3) << 1);
        bf[j][2 * v]     = Bs[nW + 16 * j + fm][k0];
        bf[j][2 * v + 1] = Bs[nW + 16 * j + fm][k0 + 1];
      }
    }
#pragma unroll
    for (int i = 0; i < 2; ++i)
#pragma unroll
      for (int j = 0; j < 2; ++j)
        acc[i][j] = __builtin_amdgcn_wmma_f32_16x16x32_f16(
            false, af[i], false, bf[j], (short)0, acc[i][j], false, false);
    __syncthreads();
  }

  // ---- epilogue + store (C/D layout: m = r + 8*khi, n = fm)
#pragma unroll
  for (int j = 0; j < 2; ++j) {
    const int gn = nBase + nW + 16 * j + fm;
    const float bv = (ep >= 1 && ep <= 3) ? bias[gn] : 0.0f;
#pragma unroll
    for (int i = 0; i < 2; ++i) {
      const long rowBase = (long)(mBase + mW + 16 * i + khi * 8) * ldc + gn;
#pragma unroll
      for (int r = 0; r < 8; ++r) {
        float v = acc[i][j][r] * alpha + bv;
        if (ep == 2)      v = tanhf(v);
        else if (ep == 3) v = gelu_erf_(v);
        else if (ep == 4) v = 1.0f / (1.0f + expf(-v));
        C[rowBase + (long)r * ldc] = v;
      }
    }
  }
}

// ---------------------------------------------------------------------------
// Bi-LSTM recurrence step: z = h@Whh^T (+precomputed Zin = x@Wih^T + b),
// PyTorch gate order i,f,g,o. grid (4 col-groups, 2 dirs), 128 threads.
// Wave w computes gate w's 64-column slice via 4 WMMA column tiles.
// h double-buffered across launches; each WG owns disjoint h/c columns.
// ---------------------------------------------------------------------------
__global__ __launch_bounds__(128)
void lstm_step(const float* __restrict__ ZinFw, const float* __restrict__ ZinBw,
               const float* __restrict__ WhhFw, const float* __restrict__ WhhBw,
               float* __restrict__ hbuf,   // [parity][dir][8][256]
               float* __restrict__ cbuf,   // [dir][8][256]
               float* __restrict__ enc,    // [B, T, 512]
               int t, int T)
{
  __shared__ _Float16 As[16][264];
  __shared__ float    zs[8][256];

  const int tid  = threadIdx.x;
  const int lane = tid & 31;
  const int wave = tid >> 5;       // gate index 0..3 (i,f,g,o)
  const int wg   = blockIdx.x;     // h column group 0..3
  const int dir  = blockIdx.y;

  const float* Zin = dir ? ZinBw : ZinFw;
  const float* Whh = dir ? WhhBw : WhhFw;
  const int ti = dir ? (T - 1 - t) : t;

  const float* hprev = hbuf + (((t & 1) * 2 + dir) * (8 * 256));
  float* hnext = hbuf + ((((t + 1) & 1) * 2 + dir) * (8 * 256));
  float* cst   = cbuf + dir * (8 * 256);

  // stage h_prev [8,256] as f16 A-tile (b128 loads), rows 8..15 zero
  {
    int row = tid >> 4;            // 0..7
    int cs  = (tid & 15) * 16;
    const float4* hp = (const float4*)(hprev + row * 256 + cs);
#pragma unroll
    for (int i = 0; i < 4; ++i) {
      float4 f = hp[i];
      h2 p0; p0.x = (_Float16)f.x; p0.y = (_Float16)f.y;
      h2 p1; p1.x = (_Float16)f.z; p1.y = (_Float16)f.w;
      *(h2*)&As[row][cs + 4 * i]     = p0;
      *(h2*)&As[row][cs + 4 * i + 2] = p1;
    }
    h2 zz; zz.x = (_Float16)0.0f; zz.y = (_Float16)0.0f;
#pragma unroll
    for (int i = 0; i < 8; ++i)
      *(h2*)&As[8 + row][cs + 2 * i] = zz;
  }
  __syncthreads();

  const int fm  = lane & 15;
  const int khi = lane >> 4;
  const int gateBase = wave * 256 + wg * 64;   // global z column base for this wave

#pragma unroll
  for (int tIdx = 0; tIdx < 4; ++tIdx) {
    const int colBase = gateBase + tIdx * 16;
    const int j = colBase + fm;                // Whh row (output gate column)
    v8f acc = {};
    for (int kc = 0; kc < 256; kc += 32) {
      v16h af, bf;
#pragma unroll
      for (int v = 0; v < 8; ++v) {
        int k0 = ((v >> 2) << 4) + (khi << 3) + ((v & 3) << 1);
        af[2 * v]     = As[fm][kc + k0];
        af[2 * v + 1] = As[fm][kc + k0 + 1];
      }
#pragma unroll
      for (int v = 0; v < 8; ++v) {
        int kk = kc + ((v >> 2) << 4) + (khi << 3) + ((v & 3) << 1);
        const float2 wv = *(const float2*)(Whh + (long)j * 256 + kk);
        bf[2 * v]     = (_Float16)wv.x;
        bf[2 * v + 1] = (_Float16)wv.y;
      }
      acc = __builtin_amdgcn_wmma_f32_16x16x32_f16(
          false, af, false, bf, (short)0, acc, false, false);
    }
    // spill z (+Zin) to LDS; only batch rows 0..7 are valid (khi==0 lanes)
    if (khi == 0) {
#pragma unroll
      for (int r = 0; r < 8; ++r) {
        int col = colBase + fm;
        float zv = acc[r] + Zin[((long)r * T + ti) * 1024 + col];
        zs[r][wave * 64 + tIdx * 16 + fm] = zv;
      }
    }
  }
  __syncthreads();

  // gate nonlinearity + state update: 8 batch x 64 cols = 512 values
#pragma unroll
  for (int u = 0; u < 4; ++u) {
    int lin = tid + 128 * u;
    int b   = lin >> 6;
    int cl  = lin & 63;
    int col = wg * 64 + cl;
    float iv = zs[b][0 * 64 + cl];
    float fv = zs[b][1 * 64 + cl];
    float gv = zs[b][2 * 64 + cl];
    float ov = zs[b][3 * 64 + cl];
    float co = cst[b * 256 + col];
    float cn = sigm_(fv) * co + sigm_(iv) * tanhf(gv);
    float hn = sigm_(ov) * tanhf(cn);
    cst[b * 256 + col]   = cn;
    hnext[b * 256 + col] = hn;
    enc[((long)b * T + ti) * 512 + dir * 256 + col] = hn;
  }
}

// ---------------------------------------------------------------------------
// small helper kernels
// ---------------------------------------------------------------------------
__global__ void fill_kernel(float* __restrict__ p, long n, float v)
{
  long i = (long)blockIdx.x * blockDim.x + threadIdx.x;
  long st = (long)gridDim.x * blockDim.x;
  for (; i < n; i += st) p[i] = v;
}

// out[row] = dot(X[row,:], w) + b[0]   (one wave32 per row)
__global__ __launch_bounds__(128)
void gemv_kernel(const float* __restrict__ X, const float* __restrict__ w,
                 const float* __restrict__ b, float* __restrict__ out,
                 int rows, int K)
{
  int row  = blockIdx.x * 4 + (threadIdx.x >> 5);
  int lane = threadIdx.x & 31;
  if (row >= rows) return;
  float s = 0.0f;
  for (int k = lane; k < K; k += 32) s += X[(long)row * K + k] * w[k];
#pragma unroll
  for (int off = 16; off >= 1; off >>= 1) s += __shfl_xor(s, off, 32);
  if (lane == 0) out[row] = s + b[0];
}

// softmax over T per block (sequence attention)
__global__ __launch_bounds__(256)
void seq_softmax(float* __restrict__ s, int T)
{
  __shared__ float red[256];
  float* sp = s + (long)blockIdx.x * T;
  float mx = -1e30f;
  for (int t = threadIdx.x; t < T; t += 256) mx = fmaxf(mx, sp[t]);
  red[threadIdx.x] = mx; __syncthreads();
  for (int st = 128; st >= 1; st >>= 1) {
    if (threadIdx.x < st) red[threadIdx.x] = fmaxf(red[threadIdx.x], red[threadIdx.x + st]);
    __syncthreads();
  }
  mx = red[0]; __syncthreads();
  float sum = 0.0f;
  for (int t = threadIdx.x; t < T; t += 256) sum += expf(sp[t] - mx);
  red[threadIdx.x] = sum; __syncthreads();
  for (int st = 128; st >= 1; st >>= 1) {
    if (threadIdx.x < st) red[threadIdx.x] += red[threadIdx.x + st];
    __syncthreads();
  }
  float inv = 1.0f / red[0]; __syncthreads();
  for (int t = threadIdx.x; t < T; t += 256) sp[t] = expf(sp[t] - mx) * inv;
}

// row softmax, one block per row of length L (attention probs, in place)
__global__ __launch_bounds__(256)
void row_softmax(float* __restrict__ S, int L)
{
  __shared__ float red[256];
  float* sp = S + (long)blockIdx.x * L;
  float mx = -1e30f;
  for (int i = threadIdx.x; i < L; i += 256) mx = fmaxf(mx, sp[i]);
  red[threadIdx.x] = mx; __syncthreads();
  for (int st = 128; st >= 1; st >>= 1) {
    if (threadIdx.x < st) red[threadIdx.x] = fmaxf(red[threadIdx.x], red[threadIdx.x + st]);
    __syncthreads();
  }
  mx = red[0]; __syncthreads();
  float sum = 0.0f;
  for (int i = threadIdx.x; i < L; i += 256) sum += expf(sp[i] - mx);
  red[threadIdx.x] = sum; __syncthreads();
  for (int st = 128; st >= 1; st >>= 1) {
    if (threadIdx.x < st) red[threadIdx.x] += red[threadIdx.x + st];
    __syncthreads();
  }
  float inv = 1.0f / red[0]; __syncthreads();
  for (int i = threadIdx.x; i < L; i += 256) sp[i] = expf(sp[i] - mx) * inv;
}

// out[i] = x[i] * w[i / D]
__global__ void scale_rows(const float* __restrict__ x, const float* __restrict__ w,
                           float* __restrict__ out, long n, int D)
{
  long i = (long)blockIdx.x * blockDim.x + threadIdx.x;
  long st = (long)gridDim.x * blockDim.x;
  for (; i < n; i += st) out[i] = x[i] * w[i / D];
}

__global__ void deg_kernel(const int* __restrict__ dst, float* __restrict__ deg, int E)
{
  int e = blockIdx.x * blockDim.x + threadIdx.x;
  if (e < E) atomicAdd(&deg[dst[e]], 1.0f);
}

// one block per edge: agg[dst] += h[src] * rsqrt(deg[src])*rsqrt(deg[dst])
__global__ __launch_bounds__(256)
void scatter_kernel(const int* __restrict__ src, const int* __restrict__ dst,
                    const float* __restrict__ deg, const float* __restrict__ h,
                    float* __restrict__ agg, int E)
{
  int e = blockIdx.x;
  if (e >= E) return;
  int s = src[e], d = dst[e];
  float norm = rsqrtf(deg[s]) * rsqrtf(deg[d]);
  const float* hp = h + (long)s * 512;
  float* ap = agg + (long)d * 512;
  for (int c = threadIdx.x; c < 512; c += 256)
    atomicAdd(ap + c, hp[c] * norm);
}

// z = LN(agg + xW/deg + bias); relu; (+res). one block per row of 512
__global__ __launch_bounds__(256)
void gcn_combine(const float* __restrict__ agg, const float* __restrict__ xW,
                 const float* __restrict__ deg, const float* __restrict__ bias,
                 const float* __restrict__ g, const float* __restrict__ bet,
                 const float* __restrict__ res, float* __restrict__ out)
{
  __shared__ float buf[512];
  __shared__ float rs[256], rq[256];
  long r = blockIdx.x;
  float idg = 1.0f / deg[r];
  float lsum = 0.0f, lsq = 0.0f;
  for (int c = threadIdx.x; c < 512; c += 256) {
    float v = agg[r * 512 + c] + xW[r * 512 + c] * idg + bias[c];
    buf[c] = v; lsum += v; lsq += v * v;
  }
  rs[threadIdx.x] = lsum; rq[threadIdx.x] = lsq; __syncthreads();
  for (int st = 128; st >= 1; st >>= 1) {
    if (threadIdx.x < st) { rs[threadIdx.x] += rs[threadIdx.x + st]; rq[threadIdx.x] += rq[threadIdx.x + st]; }
    __syncthreads();
  }
  float mu = rs[0] * (1.0f / 512.0f);
  float var = rq[0] * (1.0f / 512.0f) - mu * mu;
  float rstd = rsqrtf(var + 1e-5f);
  for (int c = threadIdx.x; c < 512; c += 256) {
    float v = (buf[c] - mu) * rstd * g[c] + bet[c];
    v = fmaxf(v, 0.0f);
    if (res) v += res[r * 512 + c];
    out[r * 512 + c] = v;
  }
}

// out = LN(x (+res)) over rows of width 512
__global__ __launch_bounds__(256)
void ln_kernel(const float* __restrict__ x, const float* __restrict__ res,
               const float* __restrict__ g, const float* __restrict__ bet,
               float* __restrict__ out)
{
  __shared__ float buf[512];
  __shared__ float rs[256], rq[256];
  long r = blockIdx.x;
  float lsum = 0.0f, lsq = 0.0f;
  for (int c = threadIdx.x; c < 512; c += 256) {
    float v = x[r * 512 + c];
    if (res) v += res[r * 512 + c];
    buf[c] = v; lsum += v; lsq += v * v;
  }
  rs[threadIdx.x] = lsum; rq[threadIdx.x] = lsq; __syncthreads();
  for (int st = 128; st >= 1; st >>= 1) {
    if (threadIdx.x < st) { rs[threadIdx.x] += rs[threadIdx.x + st]; rq[threadIdx.x] += rq[threadIdx.x + st]; }
    __syncthreads();
  }
  float mu = rs[0] * (1.0f / 512.0f);
  float var = rq[0] * (1.0f / 512.0f) - mu * mu;
  float rstd = rsqrtf(var + 1e-5f);
  for (int c = threadIdx.x; c < 512; c += 256)
    out[r * 512 + c] = (buf[c] - mu) * rstd * g[c] + bet[c];
}

__global__ void concat_kernel(const float* __restrict__ a, const float* __restrict__ b,
                              float* __restrict__ out, long rows, int d1, int d2)
{
  long n = rows * (d1 + d2);
  long i = (long)blockIdx.x * blockDim.x + threadIdx.x;
  long st = (long)gridDim.x * blockDim.x;
  for (; i < n; i += st) {
    long r = i / (d1 + d2);
    int  c = (int)(i - r * (d1 + d2));
    out[i] = (c < d1) ? a[r * d1 + c] : b[r * d2 + (c - d1)];
  }
}

// ---------------------------------------------------------------------------
// host orchestration
// ---------------------------------------------------------------------------
static inline void launch_gemm(hipStream_t st, const float* A, const float* B,
                               const float* bias, float* C,
                               int M, int N, int K, int lda, int ldb, int ldc,
                               long sAb, long sAh, long sBb, long sBh, long sCb, long sCh,
                               int batch, int nh, int transB, float alpha, int ep)
{
  dim3 g((N + 63) / 64, (M + 63) / 64, batch * nh);
  gemm_f16wmma<<<g, 128, 0, st>>>(A, B, bias, C, M, N, K, lda, ldb, ldc,
                                  sAb, sAh, sBb, sBh, sCb, sCh, nh, transB, alpha, ep);
}

static inline void launch_fill(hipStream_t st, float* p, long n, float v)
{
  int g = (int)((n + 255) / 256); if (g > 16384) g = 16384;
  fill_kernel<<<g, 256, 0, st>>>(p, n, v);
}

extern "C" void kernel_launch(void* const* d_in, const int* in_sizes, int n_in,
                              void* d_out, int out_size, void* d_ws, size_t ws_size,
                              hipStream_t stream)
{
  if (n_in < 70) return;
  const float* x_b  = (const float*)d_in[0];
  const int*   ei_b = (const int*)d_in[1];
  const float* x_g  = (const float*)d_in[3];
  const int*   ei_g = (const int*)d_in[4];
  const float* P[70];
  for (int i = 0; i < 70; ++i) P[i] = (const float*)d_in[i];

  const int B = 8, NB = 256, NG = 1024;
  const int EB = 40960, EG = 163840;
  const int RB = B * NB;   // 2048
  const int RG = B * NG;   // 8192
  const long M1 = 1048576;
  const float ascale = 0.088388347648318447f;   // 1/sqrt(128)

  float* W = (float*)d_ws;
  long off = 0;
  auto alloc = [&](long n) { float* p = W + off; off += n; return p; };

  float* enc_b = alloc((long)RB * 512);
  float* enc_g = alloc((long)RG * 512);
  float* seq_b = alloc((long)RB * 512);
  float* seq_g = alloc((long)RG * 512);
  float* str_b = alloc((long)RB * 512);
  float* str_g = alloc((long)RG * 512);
  float* fb    = alloc((long)RB * 512);
  float* fg    = alloc((long)RG * 512);
  float* b2o   = alloc((long)RB * 512);
  float* g2o   = alloc((long)RG * 512);
  float* hbuf  = alloc(2 * 2 * 8 * 256);
  float* cbuf  = alloc(2 * 8 * 256);
  float* degb  = alloc(RB);
  float* degg  = alloc(RG);
  float* svec  = alloc(RG);
  float* arena = alloc(26 * M1);
  (void)ws_size; (void)in_sizes; (void)out_size;

  // ========== 1) LSTM antibody (T=256, F=1280) ==========
  {
    float* ZinF = arena;
    float* ZinB = arena + (long)RB * 1024;
    launch_gemm(stream, x_b, P[6],  P[8],  ZinF, RB, 1024, 1280, 1280, 1280, 1024,
                0,0,0,0,0,0, 1,1, 1, 1.0f, 1);
    launch_gemm(stream, x_b, P[9],  P[11], ZinB, RB, 1024, 1280, 1280, 1280, 1024,
                0,0,0,0,0,0, 1,1, 1, 1.0f, 1);
    launch_fill(stream, hbuf, 2 * 2 * 8 * 256, 0.0f);
    launch_fill(stream, cbuf, 2 * 8 * 256, 0.0f);
    for (int t = 0; t < NB; ++t)
      lstm_step<<<dim3(4, 2), 128, 0, stream>>>(ZinF, ZinB, P[7], P[10],
                                                hbuf, cbuf, enc_b, t, NB);
  }
  // ========== 1b) LSTM antigen (T=1024) ==========
  {
    float* ZinF = arena;
    float* ZinB = arena + (long)RG * 1024;
    launch_gemm(stream, x_g, P[12], P[14], ZinF, RG, 1024, 1280, 1280, 1280, 1024,
                0,0,0,0,0,0, 1,1, 1, 1.0f, 1);
    launch_gemm(stream, x_g, P[15], P[17], ZinB, RG, 1024, 1280, 1280, 1280, 1024,
                0,0,0,0,0,0, 1,1, 1, 1.0f, 1);
    launch_fill(stream, hbuf, 2 * 2 * 8 * 256, 0.0f);
    launch_fill(stream, cbuf, 2 * 8 * 256, 0.0f);
    for (int t = 0; t < NG; ++t)
      lstm_step<<<dim3(4, 2), 128, 0, stream>>>(ZinF, ZinB, P[13], P[16],
                                                hbuf, cbuf, enc_g, t, NG);
  }

  // ========== 2) sequence attention ==========
  {
    float* t1 = arena;
    launch_gemm(stream, enc_b, P[18], P[19], t1, RB, 512, 512, 512, 512, 512,
                0,0,0,0,0,0, 1,1, 0, 1.0f, 2);
    gemv_kernel<<<RB / 4, 128, 0, stream>>>(t1, P[20], P[21], svec, RB, 512);
    seq_softmax<<<B, 256, 0, stream>>>(svec, NB);
    scale_rows<<<4096, 256, 0, stream>>>(enc_b, svec, seq_b, (long)RB * 512, 512);

    launch_gemm(stream, enc_g, P[22], P[23], t1, RG, 512, 512, 512, 512, 512,
                0,0,0,0,0,0, 1,1, 0, 1.0f, 2);
    gemv_kernel<<<RG / 4, 128, 0, stream>>>(t1, P[24], P[25], svec, RG, 512);
    seq_softmax<<<B, 256, 0, stream>>>(svec, NG);
    scale_rows<<<16384, 256, 0, stream>>>(enc_g, svec, seq_g, (long)RG * 512, 512);
  }

  // ========== 3) GCN stacks ==========
  {
    float* xW  = arena;
    float* agg = arena + 4 * M1;
    // antibody
    launch_gemm(stream, x_b, P[26], nullptr, xW, RB, 512, 1280, 1280, 512, 512,
                0,0,0,0,0,0, 1,1, 0, 1.0f, 0);
    launch_fill(stream, degb, RB, 1.0f);
    deg_kernel<<<(EB + 255) / 256, 256, 0, stream>>>(ei_b + EB, degb, EB);
    launch_fill(stream, agg, (long)RB * 512, 0.0f);
    scatter_kernel<<<EB, 256, 0, stream>>>(ei_b, ei_b + EB, degb, xW, agg, EB);
    gcn_combine<<<RB, 256, 0, stream>>>(agg, xW, degb, P[27], P[28], P[29], nullptr, str_b);
    launch_gemm(stream, str_b, P[30], nullptr, xW, RB, 512, 512, 512, 512, 512,
                0,0,0,0,0,0, 1,1, 0, 1.0f, 0);
    launch_fill(stream, agg, (long)RB * 512, 0.0f);
    scatter_kernel<<<EB, 256, 0, stream>>>(ei_b, ei_b + EB, degb, xW, agg, EB);
    gcn_combine<<<RB, 256, 0, stream>>>(agg, xW, degb, P[31], P[32], P[33], str_b, str_b);
    // antigen
    launch_gemm(stream, x_g, P[34], nullptr, xW, RG, 512, 1280, 1280, 512, 512,
                0,0,0,0,0,0, 1,1, 0, 1.0f, 0);
    launch_fill(stream, degg, RG, 1.0f);
    deg_kernel<<<(EG + 255) / 256, 256, 0, stream>>>(ei_g + EG, degg, EG);
    launch_fill(stream, agg, (long)RG * 512, 0.0f);
    scatter_kernel<<<EG, 256, 0, stream>>>(ei_g, ei_g + EG, degg, xW, agg, EG);
    gcn_combine<<<RG, 256, 0, stream>>>(agg, xW, degg, P[35], P[36], P[37], nullptr, str_g);
    launch_gemm(stream, str_g, P[38], nullptr, xW, RG, 512, 512, 512, 512, 512,
                0,0,0,0,0,0, 1,1, 0, 1.0f, 0);
    launch_fill(stream, agg, (long)RG * 512, 0.0f);
    scatter_kernel<<<EG, 256, 0, stream>>>(ei_g, ei_g + EG, degg, xW, agg, EG);
    gcn_combine<<<RG, 256, 0, stream>>>(agg, xW, degg, P[39], P[40], P[41], str_g, str_g);
  }

  // ========== 4) fuse ==========
  {
    float* cat = arena;
    float* tmp = arena + 2 * M1;
    concat_kernel<<<8192, 256, 0, stream>>>(seq_b, str_b, cat, RB, 512, 512);
    launch_gemm(stream, cat, P[42], P[43], tmp, RB, 512, 1024, 1024, 512, 512,
                0,0,0,0,0,0, 1,1, 0, 1.0f, 3);
    ln_kernel<<<RB, 256, 0, stream>>>(tmp, nullptr, P[44], P[45], fb);

    float* catg = arena;
    float* tmpg = arena + 8 * M1;
    concat_kernel<<<16384, 256, 0, stream>>>(seq_g, str_g, catg, RG, 512, 512);
    launch_gemm(stream, catg, P[46], P[47], tmpg, RG, 512, 1024, 1024, 512, 512,
                0,0,0,0,0,0, 1,1, 0, 1.0f, 3);
    ln_kernel<<<RG, 256, 0, stream>>>(tmpg, nullptr, P[48], P[49], fg);
  }

  // ========== 5) cross-attention block ==========
  {
    // --- direction 1: antibody queries antigen
    float* qb = arena;
    float* kg = arena + 1 * M1;
    float* vg = arena + 5 * M1;
    float* S  = arena + 9 * M1;    // [B,H,256,1024]
    float* Ob = arena + 17 * M1;
    float* ab = arena + 18 * M1;
    launch_gemm(stream, fb, P[50], nullptr, qb, RB, 512, 512, 512, 512, 512,
                0,0,0,0,0,0, 1,1, 0, 1.0f, 0);
    launch_gemm(stream, fg, P[51], nullptr, kg, RG, 512, 512, 512, 512, 512,
                0,0,0,0,0,0, 1,1, 0, 1.0f, 0);
    launch_gemm(stream, fg, P[52], nullptr, vg, RG, 512, 512, 512, 512, 512,
                0,0,0,0,0,0, 1,1, 0, 1.0f, 0);
    launch_gemm(stream, qb, kg, nullptr, S, 256, 1024, 128, 512, 512, 1024,
                131072, 128, 524288, 128, 1048576, 262144, 8, 4, 1, ascale, 0);
    row_softmax<<<8 * 4 * 256, 256, 0, stream>>>(S, 1024);
    launch_gemm(stream, S, vg, nullptr, Ob, 256, 128, 1024, 1024, 512, 512,
                1048576, 262144, 524288, 128, 131072, 128, 8, 4, 0, 1.0f, 0);
    launch_gemm(stream, Ob, P[53], nullptr, ab, RB, 512, 512, 512, 512, 512,
                0,0,0,0,0,0, 1,1, 0, 1.0f, 0);
    float* cat2 = arena + 19 * M1;
    float* mid  = arena + 21 * M1;
    float* fout = arena + 23 * M1;
    concat_kernel<<<8192, 256, 0, stream>>>(fb, ab, cat2, RB, 512, 512);
    launch_gemm(stream, cat2, P[58], P[59], mid, RB, 1024, 1024, 1024, 1024, 1024,
                0,0,0,0,0,0, 1,1, 0, 1.0f, 3);
    launch_gemm(stream, mid, P[60], P[61], fout, RB, 512, 1024, 1024, 512, 512,
                0,0,0,0,0,0, 1,1, 0, 1.0f, 1);
    ln_kernel<<<RB, 256, 0, stream>>>(fout, fb, P[66], P[67], b2o);

    // --- direction 2: antigen queries antibody
    float* qg = arena;
    float* kb = arena + 4 * M1;
    float* vb = arena + 5 * M1;
    float* S2 = arena + 6 * M1;    // [B,H,1024,256]
    float* Og = arena + 14 * M1;
    float* ag = arena + 18 * M1;
    launch_gemm(stream, fg, P[54], nullptr, qg, RG, 512, 512, 512, 512, 512,
                0,0,0,0,0,0, 1,1, 0, 1.0f, 0);
    launch_gemm(stream, fb, P[55], nullptr, kb, RB, 512, 512, 512, 512, 512,
                0,0,0,0,0,0, 1,1, 0, 1.0f, 0);
    launch_gemm(stream, fb, P[56], nullptr, vb, RB, 512, 512, 512, 512, 512,
                0,0,0,0,0,0, 1,1, 0, 1.0f, 0);
    launch_gemm(stream, qg, kb, nullptr, S2, 1024, 256, 128, 512, 512, 256,
                524288, 128, 131072, 128, 1048576, 262144, 8, 4, 1, ascale, 0);
    row_softmax<<<8 * 4 * 1024, 256, 0, stream>>>(S2, 256);
    launch_gemm(stream, S2, vb, nullptr, Og, 1024, 128, 256, 256, 512, 512,
                1048576, 262144, 131072, 128, 524288, 128, 8, 4, 0, 1.0f, 0);
    launch_gemm(stream, Og, P[57], nullptr, ag, RG, 512, 512, 512, 512, 512,
                0,0,0,0,0,0, 1,1, 0, 1.0f, 0);
    float* cat2g = arena;             // qg..S2..Og dead by now
    float* midg  = arena + 8 * M1;
    float* foutg = arena + 16 * M1;   // ag (18M..22M) consumed by concat first
    concat_kernel<<<16384, 256, 0, stream>>>(fg, ag, cat2g, RG, 512, 512);
    launch_gemm(stream, cat2g, P[62], P[63], midg, RG, 1024, 1024, 1024, 1024, 1024,
                0,0,0,0,0,0, 1,1, 0, 1.0f, 3);
    launch_gemm(stream, midg, P[64], P[65], foutg, RG, 512, 1024, 1024, 512, 512,
                0,0,0,0,0,0, 1,1, 0, 1.0f, 1);
    ln_kernel<<<RG, 256, 0, stream>>>(foutg, fg, P[68], P[69], g2o);
  }

  // ========== 6) inner-product decoder, sigmoid ==========
  launch_gemm(stream, b2o, g2o, nullptr, (float*)d_out, 256, 1024, 512, 512, 512, 1024,
              131072, 0, 524288, 0, 262144, 0, 8, 1, 1, 1.0f, 4);
}
